// GDN_70282844832165
// MI455X (gfx1250) — matrix operand order repaired and checked by reading, compiled, not run on hardware
//
#include <hip/hip_runtime.h>
#include <math.h>

// ---- problem dims (match reference) ----
#define B_     256
#define NODE_  512
#define WIN_   64
#define EMB_   64
#define TOPK_  16
#define INTER_ 512
#define N_     (B_ * NODE_)      // 131072 total nodes
#define NEG_   0.2f
#define EPS_   1e-5f

typedef __attribute__((ext_vector_type(16))) _Float16 v16h;
typedef __attribute__((ext_vector_type(8)))  float    v8f;

// =====================================================================
// WMMA fragment helpers (CDNA5 16x16x32 f16 -> f32, wave32; A/B use
// contiguous k = 16*(lane/16) + s, not the ISA table order).
// =====================================================================
__device__ __forceinline__ v16h load_a16x32(const float* __restrict__ A,
                                            int row0, int ld, int k0, int lane) {
  int m  = lane & 15;
  int hi = (lane >> 4) & 1;      // lanes 16-31 hold K 16..31
  const float* rp = A + (size_t)(row0 + m) * ld + k0;
  v16h a;
#pragma unroll
  for (int v = 0; v < 8; ++v) {
    int kb = hi * 16 + v * 2;
    a[2 * v]     = (_Float16)rp[kb];
    a[2 * v + 1] = (_Float16)rp[kb + 1];
  }
  return a;
}

// pre-packed B fragments: P[((tile*2 + kh)*32 + lane)*16 + e]
__device__ __forceinline__ v16h load_b_packed(const _Float16* __restrict__ P,
                                              int tile, int kh, int lane) {
  return *(const v16h*)(P + ((((size_t)tile * 2) + kh) * 32 + lane) * 16);
}

__device__ __forceinline__ v16h lds_b_frag(const _Float16* sb, int tt, int kh,
                                           int lane) {
  return *(const v16h*)(sb + (((tt * 2) + kh) * 32 + lane) * 16);
}

__device__ __forceinline__ v8f wmma_f16(v16h a, v16h b, v8f c) {
  return __builtin_amdgcn_wmma_f32_16x16x32_f16(false, a, false, b, (short)0, c,
                                                false, false);
}

__device__ __forceinline__ void store_c16x16(float* __restrict__ D, int row0,
                                             int col0, int ld, v8f c, int lane) {
  int n  = lane & 15;
  int mo = ((lane >> 4) & 1) * 8;
  float* p = D + (size_t)(row0 + mo) * ld + col0 + n;
#pragma unroll
  for (int r = 0; r < 8; ++r) p[(size_t)r * ld] = c[r];
}

// =====================================================================
// CDNA5 async global->LDS copy (ASYNCcnt path, ISA 15.18.3 op 98).
// LDS offset VGPR: low 32 bits of a flat shared-memory address ARE the
// LDS byte offset (ISA 10.2 aperture mapping), so plain truncation works.
// =====================================================================
__device__ __forceinline__ void async_cp16(unsigned lds_off, const void* gsrc) {
  asm volatile("global_load_async_to_lds_b128 %0, %1, off"
               :: "v"(lds_off), "v"(gsrc)
               : "memory");
}
__device__ __forceinline__ void async_wait0() {
  asm volatile("s_wait_asynccnt 0" ::: "memory");
}

// =====================================================================
// 0) zero stats accumulators (graph-replay safe re-init each launch)
// =====================================================================
__global__ void k_zero(float* __restrict__ p, int n) {
  int i = blockIdx.x * blockDim.x + threadIdx.x;
  if (i < n) p[i] = 0.f;
}

// =====================================================================
// 1) pack a row-major [ncols x 64] fp32 weight matrix into f16 WMMA
//    B-fragments: B[k][n] = W[(col0+n)*64 + k]. One thread per
//    (tile, k-half, lane) writes its 16 contiguous halfs.
// =====================================================================
__global__ void k_pack_b(const float* __restrict__ W, _Float16* __restrict__ P,
                         int ncols) {
  int id = blockIdx.x * blockDim.x + threadIdx.x;
  int total = (ncols >> 4) * 64;             // tiles * 2 khalves * 32 lanes
  if (id >= total) return;
  int lane = id & 31;
  int kh   = (id >> 5) & 1;
  int tile = id >> 6;
  int n  = lane & 15;
  int hi = (lane >> 4) & 1;
  const float* cp = W + (size_t)(tile * 16 + n) * 64 + kh * 32 + hi * 16;
  _Float16* o = P + (size_t)id * 16;
#pragma unroll
  for (int e = 0; e < 16; ++e) o[e] = (_Float16)cp[e];
}

// =====================================================================
// 2) per-row inverse norms of W_emb
// =====================================================================
__global__ void k_rownorm(const float* __restrict__ W, float* __restrict__ rnorm) {
  int i = blockIdx.x * blockDim.x + threadIdx.x;
  if (i >= NODE_) return;
  float s = 0.f;
#pragma unroll
  for (int k = 0; k < EMB_; ++k) { float w = W[i * EMB_ + k]; s += w * w; }
  rnorm[i] = rsqrtf(s);
}

// =====================================================================
// 3) cosine-similarity top-k (one block per row; 16 argmax sweeps,
//    tie -> lowest index, matching jax.lax.top_k's stable order)
// =====================================================================
__global__ void k_topk(const float* __restrict__ W, const float* __restrict__ rnorm,
                       int* __restrict__ topk) {
  __shared__ float cosr[NODE_];
  __shared__ float rv[NODE_];
  __shared__ int   ri[NODE_];
  int i = blockIdx.x, j = threadIdx.x;
  float s = 0.f;
#pragma unroll
  for (int k = 0; k < EMB_; ++k) s += W[i * EMB_ + k] * W[j * EMB_ + k];
  cosr[j] = s * rnorm[i] * rnorm[j];
  __syncthreads();
  for (int e = 0; e < TOPK_; ++e) {
    rv[j] = cosr[j];
    ri[j] = j;
    __syncthreads();
    for (int off = NODE_ / 2; off > 0; off >>= 1) {
      if (j < off) {
        float v2 = rv[j + off]; int i2 = ri[j + off];
        if (v2 > rv[j] || (v2 == rv[j] && i2 < ri[j])) { rv[j] = v2; ri[j] = i2; }
      }
      __syncthreads();
    }
    if (j == 0) { topk[i * TOPK_ + e] = ri[0]; cosr[ri[0]] = -INFINITY; }
    __syncthreads();
  }
}

// =====================================================================
// 4) xl = x @ W_lin^T  (WMMA, pre-packed f16 B fragments)
// =====================================================================
__global__ void k_gemm_xl(const float* __restrict__ X,
                          const _Float16* __restrict__ Bp,
                          float* __restrict__ XL) {
  int lane = threadIdx.x & 31, wave = threadIdx.x >> 5;
  int row0 = (blockIdx.x * 8 + wave) * 16;
  v16h a0 = load_a16x32(X, row0, WIN_, 0, lane);
  v16h a1 = load_a16x32(X, row0, WIN_, 32, lane);
#pragma unroll
  for (int t = 0; t < EMB_ / 16; ++t) {
    v16h b0 = load_b_packed(Bp, t, 0, lane);
    v16h b1 = load_b_packed(Bp, t, 1, lane);
    v8f acc = {};
    acc = wmma_f16(a0, b0, acc);
    acc = wmma_f16(a1, b1, acc);
    store_c16x16(XL, row0, t * 16, EMB_, acc, lane);
  }
}

// =====================================================================
// 5) node attention scores s_i/s_j (includes tiled embedding terms)
// =====================================================================
__global__ void k_scores(const float* __restrict__ xl, const float* __restrict__ Wemb,
                         const float* __restrict__ ai, const float* __restrict__ aj,
                         const float* __restrict__ aei, const float* __restrict__ aej,
                         float* __restrict__ s_i, float* __restrict__ s_j) {
  int n = blockIdx.x * blockDim.x + threadIdx.x;
  if (n >= N_) return;
  const float* xr = xl + (size_t)n * EMB_;
  const float* er = Wemb + (size_t)(n & (NODE_ - 1)) * EMB_;
  float si = 0.f, sj = 0.f;
#pragma unroll
  for (int k = 0; k < EMB_; ++k) {
    si += xr[k] * ai[k] + er[k] * aei[k];
    sj += xr[k] * aj[k] + er[k] * aej[k];
  }
  s_i[n] = si;
  s_j[n] = sj;
}

// =====================================================================
// 6) edge softmax + aggregation (TOPK contiguous edges per dst node)
// =====================================================================
__global__ void k_edge_agg(const float* __restrict__ xl, const float* __restrict__ s_i,
                           const float* __restrict__ s_j, const int* __restrict__ topk,
                           float* __restrict__ agg) {
  int t = threadIdx.x & 63;
  int n = blockIdx.x * 4 + (threadIdx.x >> 6);
  int node = n & (NODE_ - 1);
  int boff = n - node;
  float si = s_i[n];
  float al[TOPK_];
  int   src[TOPK_];
  float amax = -INFINITY;
#pragma unroll
  for (int e = 0; e < TOPK_; ++e) {
    int s = topk[node * TOPK_ + e] + boff;
    src[e] = s;
    float a = si + s_j[s];
    a = (a > 0.f) ? a : NEG_ * a;
    al[e] = a;
    amax = fmaxf(amax, a);
  }
  float den = 0.f;
#pragma unroll
  for (int e = 0; e < TOPK_; ++e) { al[e] = __expf(al[e] - amax); den += al[e]; }
  float inv = 1.f / den;
  float acc = 0.f;
#pragma unroll
  for (int e = 0; e < TOPK_; ++e)
    acc += al[e] * inv * xl[(size_t)src[e] * EMB_ + t];
  agg[(size_t)n * EMB_ + t] = acc;
}

// =====================================================================
// 7) per-channel sum / sumsq for a [N,64] tensor
// =====================================================================
__global__ void k_bn_accum64(const float* __restrict__ x, float* __restrict__ gsum,
                             float* __restrict__ gsq) {
  __shared__ float ls[EMB_], lq[EMB_];
  int t = threadIdx.x, c = t & 63;
  if (t < EMB_) { ls[t] = 0.f; lq[t] = 0.f; }
  __syncthreads();
  size_t base = (size_t)blockIdx.x * 16384;
  float s = 0.f, q = 0.f;
#pragma unroll 4
  for (int k = 0; k < 64; ++k) {
    float v = x[base + (size_t)k * 256 + t];
    s += v; q += v * v;
  }
  atomicAdd(&ls[c], s);
  atomicAdd(&lq[c], q);
  __syncthreads();
  if (t < EMB_) { atomicAdd(&gsum[t], ls[t]); atomicAdd(&gsq[t], lq[t]); }
}

// =====================================================================
// 8) finalize BN stats: mean, rsqrt(var+eps) (biased var)
// =====================================================================
__global__ void k_bn_final(const float* __restrict__ sum, const float* __restrict__ sq,
                           float* __restrict__ mean, float* __restrict__ rsig,
                           int C, float invN) {
  int c = blockIdx.x * blockDim.x + threadIdx.x;
  if (c >= C) return;
  float m = sum[c] * invN;
  mean[c] = m;
  rsig[c] = rsqrtf(sq[c] * invN - m * m + EPS_);
}

// =====================================================================
// 9) z = relu(bn1(agg)) * W_emb_tiled (+ bn2 stats).  b_gnn cancels
//    exactly under training-mode BN -> dropped.
// =====================================================================
__global__ void k_apply1(const float* __restrict__ agg, const float* __restrict__ mean1,
                         const float* __restrict__ rsig1, const float* __restrict__ g1,
                         const float* __restrict__ be1, const float* __restrict__ Wemb,
                         float* __restrict__ z, float* __restrict__ gsum2,
                         float* __restrict__ gsq2) {
  __shared__ float ls[EMB_], lq[EMB_];
  int t = threadIdx.x, c = t & 63;
  if (t < EMB_) { ls[t] = 0.f; lq[t] = 0.f; }
  __syncthreads();
  float sc = rsig1[c] * g1[c];
  float sb = be1[c] - mean1[c] * sc;
  size_t base = (size_t)blockIdx.x * 16384;
  float s = 0.f, q = 0.f;
#pragma unroll 4
  for (int k = 0; k < 64; ++k) {
    size_t idx = base + (size_t)k * 256 + t;
    float h = fmaxf(agg[idx] * sc + sb, 0.f);
    int n = (int)(idx >> 6);
    float zz = h * Wemb[((n & (NODE_ - 1)) << 6) + c];
    z[idx] = zz;
    s += zz; q += zz * zz;
  }
  atomicAdd(&ls[c], s);
  atomicAdd(&lq[c], q);
  __syncthreads();
  if (t < EMB_) { atomicAdd(&gsum2[t], ls[t]); atomicAdd(&gsq2[t], lq[t]); }
}

// =====================================================================
// 10) in-place y = relu(bn2(z))
// =====================================================================
__global__ void k_apply2(float* __restrict__ z, const float* __restrict__ mean2,
                         const float* __restrict__ rsig2, const float* __restrict__ g2,
                         const float* __restrict__ be2) {
  size_t idx = (size_t)blockIdx.x * blockDim.x + threadIdx.x;
  int c = (int)(idx & 63);
  float sc = rsig2[c] * g2[c];
  z[idx] = fmaxf(z[idx] * sc + (be2[c] - mean2[c] * sc), 0.f);
}

// =====================================================================
// 11) GEMM pass 1: t = y @ W1^T with per-INTER-channel sum/sumsq.
//     B fragments staged via async global->LDS in 16KB super-tiles.
//     b1 cancels exactly under training-mode BN3 -> dropped.
// =====================================================================
#define SUP_TILES 8                               // col tiles per stage
#define SUP_HALFS (SUP_TILES * 2 * 32 * 16)       // 8192 halfs = 16 KB
#define SUP_CHUNKS (SUP_HALFS / 8)                // 16B b128 chunks = 1024

__global__ void k_gemm_stats(const float* __restrict__ Y,
                             const _Float16* __restrict__ W1p,
                             float* __restrict__ gsum, float* __restrict__ gsq) {
  __shared__ __align__(64) _Float16 sb[SUP_HALFS];
  __shared__ float ls[INTER_], lq[INTER_];
  for (int i = threadIdx.x; i < INTER_; i += blockDim.x) { ls[i] = 0.f; lq[i] = 0.f; }
  int lane = threadIdx.x & 31, wave = threadIdx.x >> 5;
  int row0 = (blockIdx.x * 8 + wave) * 16;
  v16h a0 = load_a16x32(Y, row0, EMB_, 0, lane);
  v16h a1 = load_a16x32(Y, row0, EMB_, 32, lane);
  __syncthreads();
  for (int sup = 0; sup < (INTER_ / 16) / SUP_TILES; ++sup) {
    const _Float16* src = W1p + (size_t)sup * SUP_HALFS;
    for (int i = threadIdx.x; i < SUP_CHUNKS; i += 256)
      async_cp16((unsigned)(uintptr_t)&sb[i * 8], src + (size_t)i * 8);
    async_wait0();
    __syncthreads();
#pragma unroll
    for (int tt = 0; tt < SUP_TILES; ++tt) {
      v16h b0 = lds_b_frag(sb, tt, 0, lane);
      v16h b1 = lds_b_frag(sb, tt, 1, lane);
      v8f acc = {};
      acc = wmma_f16(a0, b0, acc);
      acc = wmma_f16(a1, b1, acc);
      float p = 0.f, q = 0.f;
#pragma unroll
      for (int r = 0; r < 8; ++r) { p += acc[r]; q += acc[r] * acc[r]; }
      int ch = (sup * SUP_TILES + tt) * 16 + (lane & 15);
      atomicAdd(&ls[ch], p);
      atomicAdd(&lq[ch], q);
    }
    __syncthreads();
  }
  for (int i = threadIdx.x; i < INTER_; i += blockDim.x) {
    atomicAdd(&gsum[i], ls[i]);
    atomicAdd(&gsq[i], lq[i]);
  }
}

// =====================================================================
// 12) GEMM pass 2: recompute t, bn3+relu, dot W2, + b2 -> out
// =====================================================================
__global__ void k_gemm_out(const float* __restrict__ Y,
                           const _Float16* __restrict__ W1p,
                           const float* __restrict__ mean3, const float* __restrict__ rsig3,
                           const float* __restrict__ g3, const float* __restrict__ be3,
                           const float* __restrict__ W2, const float* __restrict__ b2,
                           float* __restrict__ out) {
  __shared__ __align__(64) _Float16 sb[SUP_HALFS];
  int lane = threadIdx.x & 31, wave = threadIdx.x >> 5;
  int row0 = (blockIdx.x * 8 + wave) * 16;
  v16h a0 = load_a16x32(Y, row0, EMB_, 0, lane);
  v16h a1 = load_a16x32(Y, row0, EMB_, 32, lane);
  float racc[8] = {0.f, 0.f, 0.f, 0.f, 0.f, 0.f, 0.f, 0.f};
  for (int sup = 0; sup < (INTER_ / 16) / SUP_TILES; ++sup) {
    const _Float16* src = W1p + (size_t)sup * SUP_HALFS;
    for (int i = threadIdx.x; i < SUP_CHUNKS; i += 256)
      async_cp16((unsigned)(uintptr_t)&sb[i * 8], src + (size_t)i * 8);
    async_wait0();
    __syncthreads();
#pragma unroll
    for (int tt = 0; tt < SUP_TILES; ++tt) {
      v16h b0 = lds_b_frag(sb, tt, 0, lane);
      v16h b1 = lds_b_frag(sb, tt, 1, lane);
      v8f acc = {};
      acc = wmma_f16(a0, b0, acc);
      acc = wmma_f16(a1, b1, acc);
      int ch = (sup * SUP_TILES + tt) * 16 + (lane & 15);
      float sc = rsig3[ch] * g3[ch];
      float sbv = be3[ch] - mean3[ch] * sc;
      float w2 = W2[ch];
#pragma unroll
      for (int r = 0; r < 8; ++r) {
        float zz = fmaxf(acc[r] * sc + sbv, 0.f);
        racc[r] += zz * w2;
      }
    }
    __syncthreads();
  }
#pragma unroll
  for (int off = 1; off < 16; off <<= 1)
#pragma unroll
    for (int r = 0; r < 8; ++r) racc[r] += __shfl_xor(racc[r], off, 16);
  float bias = b2[0];
  if (lane == 0) {
#pragma unroll
    for (int r = 0; r < 8; ++r) out[row0 + r] = racc[r] + bias;
  } else if (lane == 16) {
#pragma unroll
    for (int r = 0; r < 8; ++r) out[row0 + 8 + r] = racc[r] + bias;
  }
}

// =====================================================================
// host-side launch
// =====================================================================
extern "C" void kernel_launch(void* const* d_in, const int* in_sizes, int n_in,
                              void* d_out, int out_size, void* d_ws, size_t ws_size,
                              hipStream_t stream) {
  (void)in_sizes; (void)n_in; (void)out_size; (void)ws_size;
  const float* data  = (const float*)d_in[0];
  // d_in[1] org_edge_index: unused by the forward pass
  const float* Wemb  = (const float*)d_in[2];
  const float* Wlin  = (const float*)d_in[3];
  const float* att_i = (const float*)d_in[4];
  const float* att_j = (const float*)d_in[5];
  const float* aem_i = (const float*)d_in[6];
  const float* aem_j = (const float*)d_in[7];
  // d_in[8] b_gnn: exact no-op under training-mode BN1
  const float* g1    = (const float*)d_in[9];
  const float* be1   = (const float*)d_in[10];
  const float* g2    = (const float*)d_in[11];
  const float* be2   = (const float*)d_in[12];
  const float* W1    = (const float*)d_in[13];
  // d_in[14] b1: exact no-op under training-mode BN3
  const float* g3    = (const float*)d_in[15];
  const float* be3   = (const float*)d_in[16];
  const float* W2    = (const float*)d_in[17];
  const float* b2    = (const float*)d_in[18];
  float* out = (float*)d_out;

  char* ws = (char*)d_ws;
  size_t off = 0;
  auto carve = [&](size_t bytes) {
    void* p = ws + off;
    off += (bytes + 255) & ~(size_t)255;
    return p;
  };
  float* bufA  = (float*)carve((size_t)N_ * EMB_ * 4);  // xl -> z -> y
  float* bufB  = (float*)carve((size_t)N_ * EMB_ * 4);  // agg
  float* s_i   = (float*)carve((size_t)N_ * 4);
  float* s_j   = (float*)carve((size_t)N_ * 4);
  int*   topk  = (int*)  carve((size_t)NODE_ * TOPK_ * 4);
  float* rnorm = (float*)carve((size_t)NODE_ * 4);
  _Float16* w1h  = (_Float16*)carve((size_t)INTER_ * EMB_ * 2);  // packed W1 frags
  _Float16* wlh  = (_Float16*)carve((size_t)EMB_ * WIN_ * 2);    // packed W_lin frags
  float* stats = (float*)carve((size_t)2560 * 4);
  float *sum1 = stats,        *sq1 = stats + 64;
  float *sum2 = stats + 128,  *sq2 = stats + 192;
  float *sum3 = stats + 256,  *sq3 = stats + 768;
  float *mean1 = stats + 1280, *rsig1 = stats + 1344;
  float *mean2 = stats + 1408, *rsig2 = stats + 1472;
  float *mean3 = stats + 1536, *rsig3 = stats + 2048;

  const float invN = 1.f / (float)N_;

  k_zero<<<5, 256, 0, stream>>>(stats, 1280);
  k_pack_b<<<1, 256, 0, stream>>>(Wlin, wlh, EMB_);
  k_pack_b<<<8, 256, 0, stream>>>(W1, w1h, INTER_);
  k_rownorm<<<2, 256, 0, stream>>>(Wemb, rnorm);
  k_topk<<<NODE_, NODE_, 0, stream>>>(Wemb, rnorm, topk);
  k_gemm_xl<<<N_ / 128, 256, 0, stream>>>(data, wlh, bufA);
  k_scores<<<N_ / 256, 256, 0, stream>>>(bufA, Wemb, att_i, att_j, aem_i, aem_j,
                                         s_i, s_j);
  k_edge_agg<<<N_ / 4, 256, 0, stream>>>(bufA, s_i, s_j, topk, bufB);
  k_bn_accum64<<<512, 256, 0, stream>>>(bufB, sum1, sq1);
  k_bn_final<<<1, 64, 0, stream>>>(sum1, sq1, mean1, rsig1, EMB_, invN);
  k_apply1<<<512, 256, 0, stream>>>(bufB, mean1, rsig1, g1, be1, Wemb, bufA,
                                    sum2, sq2);
  k_bn_final<<<1, 64, 0, stream>>>(sum2, sq2, mean2, rsig2, EMB_, invN);
  k_apply2<<<(N_ * EMB_) / 256, 256, 0, stream>>>(bufA, mean2, rsig2, g2, be2);
  k_gemm_stats<<<N_ / 128, 256, 0, stream>>>(bufA, w1h, sum3, sq3);
  k_bn_final<<<2, 256, 0, stream>>>(sum3, sq3, mean3, rsig3, INTER_, invN);
  k_gemm_out<<<N_ / 128, 256, 0, stream>>>(bufA, w1h, mean3, rsig3, g3, be3, W2, b2,
                                           out);
}
